// RPN_61572651155844
// MI455X (gfx1250) — matrix operand order repaired and checked by reading, not compiled, not run on hardware
//
#include <hip/hip_runtime.h>
#include <hip/hip_bf16.h>

typedef __attribute__((ext_vector_type(16))) __bf16 v16bf;
typedef __attribute__((ext_vector_type(8)))  __bf16 v8bf;
typedef __attribute__((ext_vector_type(8)))  float  v8f;

#define HH 64
#define WW 64
#define WP 66                   /* padded width/height (1-px zero halo) */
#define CC 256
#define NN 8
#define AA 9
#define HWA (HH*WW*AA)          /* 36864 */
#define PRE_NMS 2000
#define POST_NMS 1000
#define CAND_CAP 4096
#define NMS_T 0.7f
#define BBOX_CLIP 4.135166556742356f

// ---------------- pack kernels ----------------

// NCHW f32 -> padded NHWC bf16 with zero halo (removes all boundary branches
// from the WMMA conv inner loop)
__global__ void pack_feat_pad(const float* __restrict__ in, __bf16* __restrict__ out, int total) {
    int i = blockIdx.x * 256 + threadIdx.x;
    if (i >= total) return;
    int c     = i & 255;
    int rest  = i >> 8;
    int xp    = rest % WP;
    int rest2 = rest / WP;
    int yp    = rest2 % WP;
    int n     = rest2 / WP;
    float v = 0.f;
    if (xp >= 1 && xp <= WW && yp >= 1 && yp <= HH)
        v = in[(((size_t)n * CC + c) * HH + (yp - 1)) * WW + (xp - 1)];
    out[i] = (__bf16)v;
}

// conv_w [Co][Ci][3][3] -> per-tap, per-kchunk, per-ntile B fragments
// flat: ((t*8 + kc)*16 + nt)*512 + lane*16 + j
__global__ void pack_w3(const float* __restrict__ w, __bf16* __restrict__ wp, int total) {
    int i = blockIdx.x * 256 + threadIdx.x;
    if (i >= total) return;
    int j    = i & 15;
    int lane = (i >> 4) & 31;
    int nt   = (i >> 9) & 15;
    int kc   = (i >> 13) & 7;
    int t    = i >> 16;
    int K  = kc * 32 + ((lane >= 16) ? 8 : 0) + (j & 7) + ((j >> 3) << 4);   // B-matrix K mapping (ISA 7.12.2)
    int Nc = nt * 16 + (lane & 15);
    int ky = t / 3, kx = t % 3;
    wp[i] = (__bf16)w[((Nc * CC + K) * 3 + ky) * 3 + kx];
}

// head weights: cols 0..8 = cls, 9..44 = bbox, 45..47 = 0
// flat: (kc*3 + nt)*512 + lane*16 + j
__global__ void pack_whead(const float* __restrict__ cls_w, const float* __restrict__ bbox_w,
                           __bf16* __restrict__ wp, int total) {
    int i = blockIdx.x * 256 + threadIdx.x;
    if (i >= total) return;
    int j    = i & 15;
    int lane = (i >> 4) & 31;
    int blk  = i >> 9;
    int nt = blk % 3, kc = blk / 3;
    int K = kc * 32 + ((lane >= 16) ? 8 : 0) + (j & 7) + ((j >> 3) << 4);
    int c = nt * 16 + (lane & 15);
    float v = 0.f;
    if (c < 9)       v = cls_w[c * CC + K];
    else if (c < 45) v = bbox_w[(c - 9) * CC + K];
    wp[i] = (__bf16)v;
}

// ---------------- conv 3x3 as 9 shifted WMMA GEMMs (halo-padded, branch-free) ----------------

__global__ void __launch_bounds__(256) conv3x3_wmma(
        const __bf16* __restrict__ feat, const __bf16* __restrict__ wp,
        const float* __restrict__ conv_b, __bf16* __restrict__ tout) {
    const int lane = threadIdx.x & 31;
    const int wave = threadIdx.x >> 5;
    const int mi = wave >> 2;                 // 0..1
    const int ni = wave & 3;                  // 0..3
    const int pix0  = blockIdx.x * 32 + mi * 16;
    const int ntile = blockIdx.y * 4 + ni;    // 0..15
    const int co0   = ntile * 16;

    const int n   = pix0 >> 12;
    const int rem = pix0 & 4095;
    const int y   = rem >> 6;
    const int x0  = rem & 63;                 // multiple of 16 (16 | 64)
    const int mrow = lane & 15;
    const int x = x0 + mrow;
    const int abase = (lane >> 4) << 3;       // K half select for A (ISA 7.12.2)

    // Padded pixel (y+ky-1+1, x+kx-1+1) == (y+ky, x+kx); tap (0,0) base:
    const __bf16* a00   = feat + ((size_t)((n * WP + y) * WP + x)) * CC + abase;
    const __bf16* bbase = wp + (size_t)ntile * 512 + lane * 16;

    v8f acc;
#pragma unroll
    for (int r = 0; r < 8; ++r) acc[r] = 0.f;

#pragma unroll
    for (int t = 0; t < 9; ++t) {
        const int ky = t / 3, kx = t % 3;
        const __bf16* arow = a00 + (ky * WP + kx) * CC;
#pragma unroll
        for (int kc = 0; kc < 8; ++kc) {
            // unconditional loads; fully unrolled -> compile-time immediate offsets
            v8bf lo = *(const v8bf*)(arow + kc * 32);
            v8bf hi = *(const v8bf*)(arow + kc * 32 + 16);
            v16bf a;
#pragma unroll
            for (int j = 0; j < 8; ++j) { a[j] = lo[j]; a[j + 8] = hi[j]; }
            v16bf b = *(const v16bf*)(bbase + (size_t)(t * 8 + kc) * 8192);
            acc = __builtin_amdgcn_wmma_f32_16x16x32_bf16(false, a, false, b,
                                                          (short)0, acc, false, false);
        }
    }

    const int col = co0 + (lane & 15);
    const float bias = conv_b[col];
    const int rbase = (lane >> 4) << 3;       // C/D: lanes>=16 hold M=8..15
#pragma unroll
    for (int r = 0; r < 8; ++r) {
        float v = acc[r] + bias;
        v = v > 0.f ? v : 0.f;                // fused ReLU
        int pix = pix0 + rbase + r;
        tout[(size_t)pix * CC + col] = (__bf16)v;
    }
}

// ---------------- head 1x1 convs (cls + bbox) as one WMMA GEMM ----------------

__global__ void __launch_bounds__(256) head_wmma(
        const __bf16* __restrict__ tin, const __bf16* __restrict__ wp,
        const float* __restrict__ cls_b, const float* __restrict__ bbox_b,
        float* __restrict__ logits_out, float* __restrict__ obj,
        float* __restrict__ boxreg) {
    const int lane = threadIdx.x & 31;
    const int wave = threadIdx.x >> 5;
    const int pix0 = blockIdx.x * 128 + wave * 16;
    const int mrow = lane & 15;
    const int abase = (lane >> 4) << 3;
    const __bf16* arow = tin + ((size_t)(pix0 + mrow)) * CC + abase;

    v8f acc[3];
#pragma unroll
    for (int nt = 0; nt < 3; ++nt)
#pragma unroll
        for (int r = 0; r < 8; ++r) acc[nt][r] = 0.f;

#pragma unroll
    for (int kc = 0; kc < 8; ++kc) {
        v8bf lo = *(const v8bf*)(arow + kc * 32);
        v8bf hi = *(const v8bf*)(arow + kc * 32 + 16);
        v16bf a;
#pragma unroll
        for (int j = 0; j < 8; ++j) { a[j] = lo[j]; a[j + 8] = hi[j]; }
#pragma unroll
        for (int nt = 0; nt < 3; ++nt) {
            v16bf b = *(const v16bf*)(wp + ((size_t)(kc * 3 + nt)) * 512 + lane * 16);
            acc[nt] = __builtin_amdgcn_wmma_f32_16x16x32_bf16(false, a, false, b,
                                                              (short)0, acc[nt], false, false);
        }
    }

    const int rbase = (lane >> 4) << 3;
#pragma unroll
    for (int nt = 0; nt < 3; ++nt) {
        int c = nt * 16 + (lane & 15);
#pragma unroll
        for (int r = 0; r < 8; ++r) {
            int p = pix0 + rbase + r;
            int n = p >> 12;
            int cell = p & 4095;              // y*64 + x
            float v = acc[nt][r];
            if (c < 9) {
                v += cls_b[c];
                logits_out[(((size_t)n * 9 + c) << 12) + cell] = v;   // NCHW logits
                obj[(size_t)n * HWA + cell * 9 + c] = 1.f / (1.f + expf(-v));
            } else if (c < 45) {
                int q = c - 9;
                v += bbox_b[q];
                boxreg[((size_t)n * HWA + cell * 9 + (q >> 2)) * 4 + (q & 3)] = v;
            }
        }
    }
}

// ---------------- top-k via histogram threshold ----------------

__global__ void topk_select(const float* __restrict__ obj,
                            float* __restrict__ cand_s, int* __restrict__ cand_i,
                            int* __restrict__ cand_cnt) {
    const int n = blockIdx.x;
    const float* s = obj + (size_t)n * HWA;
    __shared__ int hist[1024];
    __shared__ int cutbin;
    for (int i = threadIdx.x; i < 1024; i += blockDim.x) hist[i] = 0;
    __syncthreads();
    for (int i = threadIdx.x; i < HWA; i += blockDim.x) {
        int b = (int)(s[i] * 1024.f);
        b = b < 0 ? 0 : (b > 1023 ? 1023 : b);
        atomicAdd(&hist[b], 1);
    }
    __syncthreads();
    if (threadIdx.x == 0) {
        int acc = 0, cb = 0;
        for (int b = 1023; b >= 0; --b) {
            acc += hist[b];
            if (acc >= PRE_NMS) { cb = b; break; }
        }
        cutbin = cb;
    }
    __syncthreads();
    const int cb = cutbin;
    for (int i = threadIdx.x; i < HWA; i += blockDim.x) {
        float v = s[i];
        int b = (int)(v * 1024.f);
        b = b < 0 ? 0 : (b > 1023 ? 1023 : b);
        if (b >= cb) {
            int pos = atomicAdd(&cand_cnt[n], 1);
            if (pos < CAND_CAP) {
                cand_s[(size_t)n * CAND_CAP + pos] = v;
                cand_i[(size_t)n * CAND_CAP + pos] = i;
            }
        }
    }
}

// ---------------- LDS bitonic sort (descending) ----------------

__global__ void __launch_bounds__(1024) sort_cand(float* __restrict__ cand_s,
                                                  int* __restrict__ cand_i,
                                                  const int* __restrict__ cand_cnt) {
    const int n = blockIdx.x;
    __shared__ float ks[CAND_CAP];
    __shared__ int   vs[CAND_CAP];
    int cnt = cand_cnt[n]; if (cnt > CAND_CAP) cnt = CAND_CAP;
    for (int i = threadIdx.x; i < CAND_CAP; i += blockDim.x) {
        if (i < cnt) { ks[i] = cand_s[(size_t)n * CAND_CAP + i]; vs[i] = cand_i[(size_t)n * CAND_CAP + i]; }
        else         { ks[i] = -1.f; vs[i] = 0; }
    }
    __syncthreads();
    for (unsigned k = 2; k <= CAND_CAP; k <<= 1) {
        for (unsigned j = k >> 1; j > 0; j >>= 1) {
            for (unsigned i = threadIdx.x; i < CAND_CAP; i += blockDim.x) {
                unsigned ixj = i ^ j;
                if (ixj > i) {
                    bool asc = (i & k) != 0;          // overall descending
                    float a = ks[i], b = ks[ixj];
                    if (asc ? (a > b) : (a < b)) {
                        ks[i] = b; ks[ixj] = a;
                        int t = vs[i]; vs[i] = vs[ixj]; vs[ixj] = t;
                    }
                }
            }
            __syncthreads();
        }
    }
    for (int i = threadIdx.x; i < CAND_CAP; i += blockDim.x) {
        cand_s[(size_t)n * CAND_CAP + i] = ks[i];
        cand_i[(size_t)n * CAND_CAP + i] = vs[i];
    }
}

// ---------------- decode + clip ----------------

__global__ void decode_clip(const float* __restrict__ cand_s, const int* __restrict__ cand_i,
                            const float* __restrict__ boxreg,
                            const int* __restrict__ imgh, const int* __restrict__ imgw,
                            float* __restrict__ prop, float* __restrict__ pscore,
                            int* __restrict__ pvalid) {
    int g = blockIdx.x * 256 + threadIdx.x;
    if (g >= NN * PRE_NMS) return;
    int n = g / PRE_NMS, r = g % PRE_NMS;
    float sc = cand_s[(size_t)n * CAND_CAP + r];
    int idx  = cand_i[(size_t)n * CAND_CAP + r];
    int a = idx % 9, cell = idx / 9;
    int x = cell & 63, y = cell >> 6;
    const float scales[3] = {128.f, 256.f, 512.f};
    const float ratios[3] = {0.5f, 1.f, 2.f};
    int ri = a / 3, si = a % 3;
    float hr = sqrtf(ratios[ri]);
    float wr = 1.f / hr;
    float wa = wr * scales[si], ha = hr * scales[si];
    float cx = x * 16.f, cy = y * 16.f;
    const float* rg = boxreg + ((size_t)n * HWA + idx) * 4;
    float dx = rg[0], dy = rg[1];
    float dw = fminf(rg[2], BBOX_CLIP), dh = fminf(rg[3], BBOX_CLIP);
    float pcx = dx * wa + cx, pcy = dy * ha + cy;
    float pw = expf(dw) * wa, ph = expf(dh) * ha;
    float fw = (float)imgw[0], fh = (float)imgh[0];
    float x1 = fminf(fmaxf(pcx - 0.5f * pw, 0.f), fw);
    float y1 = fminf(fmaxf(pcy - 0.5f * ph, 0.f), fh);
    float x2 = fminf(fmaxf(pcx + 0.5f * pw, 0.f), fw);
    float y2 = fminf(fmaxf(pcy + 0.5f * ph, 0.f), fh);
    float* o = prop + (size_t)g * 4;
    o[0] = x1; o[1] = y1; o[2] = x2; o[3] = y2;
    pscore[g] = sc;
    pvalid[g] = ((x2 - x1 >= 0.f) && (y2 - y1 >= 0.f)) ? 1 : 0;
}

// ---------------- greedy NMS (boxes cached in LDS) ----------------

__global__ void __launch_bounds__(256) nms_kernel(
        const float* __restrict__ prop, const float* __restrict__ pscore,
        const int* __restrict__ pvalid,
        float* __restrict__ out_props, float* __restrict__ out_scrs,
        float* __restrict__ out_mask) {
    const int n = blockIdx.x;
    __shared__ float bx[PRE_NMS][4];
    __shared__ float area[PRE_NMS];
    __shared__ unsigned char supp[PRE_NMS];
    __shared__ unsigned char keep[PRE_NMS];
    __shared__ short sel[POST_NMS];
    __shared__ int nk;
    for (int i = threadIdx.x; i < PRE_NMS; i += blockDim.x) {
        const float* p = prop + ((size_t)n * PRE_NMS + i) * 4;
        bx[i][0] = p[0]; bx[i][1] = p[1]; bx[i][2] = p[2]; bx[i][3] = p[3];
        area[i] = (p[2] - p[0]) * (p[3] - p[1]);
        supp[i] = 0; keep[i] = 0;
    }
    __syncthreads();
    for (int i = 0; i < PRE_NMS; ++i) {
        bool ki = (pvalid[(size_t)n * PRE_NMS + i] != 0) && (supp[i] == 0);
        if (threadIdx.x == 0) keep[i] = ki ? 1 : 0;
        if (ki) {
            float ax1 = bx[i][0], ay1 = bx[i][1], ax2 = bx[i][2], ay2 = bx[i][3], aa = area[i];
            for (int j = i + 1 + threadIdx.x; j < PRE_NMS; j += blockDim.x) {
                float lx = fmaxf(ax1, bx[j][0]);
                float ly = fmaxf(ay1, bx[j][1]);
                float rx = fminf(ax2, bx[j][2]);
                float ry = fminf(ay2, bx[j][3]);
                float iw = fmaxf(rx - lx, 0.f), ih = fmaxf(ry - ly, 0.f);
                float inter = iw * ih;
                float iou = inter / (aa + area[j] - inter);
                if (iou > NMS_T) supp[j] = 1;
            }
        }
        __syncthreads();
    }
    if (threadIdx.x == 0) {
        int c = 0;
        for (int i = 0; i < PRE_NMS && c < POST_NMS; ++i)
            if (keep[i]) sel[c++] = (short)i;
        nk = c;
    }
    __syncthreads();
    const int kcnt = nk;
    for (int r = threadIdx.x; r < POST_NMS; r += blockDim.x) {
        float p0 = 0, p1 = 0, p2 = 0, p3 = 0, s = 0, m = 0;
        if (r < kcnt) {
            int i = sel[r];
            p0 = bx[i][0]; p1 = bx[i][1]; p2 = bx[i][2]; p3 = bx[i][3];
            s = pscore[(size_t)n * PRE_NMS + i];
            m = 1.f;
        }
        float* op = out_props + ((size_t)n * POST_NMS + r) * 4;
        op[0] = p0; op[1] = p1; op[2] = p2; op[3] = p3;
        out_scrs[(size_t)n * POST_NMS + r] = s;
        out_mask[(size_t)n * POST_NMS + r] = m;
    }
}

// ---------------- launch ----------------

extern "C" void kernel_launch(void* const* d_in, const int* in_sizes, int n_in,
                              void* d_out, int out_size, void* d_ws, size_t ws_size,
                              hipStream_t stream) {
    const float* features = (const float*)d_in[0];
    const int*   img_h    = (const int*)d_in[1];
    const int*   img_w    = (const int*)d_in[2];
    const float* conv_w   = (const float*)d_in[3];
    const float* conv_b   = (const float*)d_in[4];
    const float* cls_w    = (const float*)d_in[5];
    const float* cls_b    = (const float*)d_in[6];
    const float* bbox_w   = (const float*)d_in[7];
    const float* bbox_b   = (const float*)d_in[8];
    (void)in_sizes; (void)n_in; (void)out_size; (void)ws_size;

    char* ws = (char*)d_ws;
    size_t off = 0;
    auto carve = [&](size_t bytes) -> void* {
        void* p = ws + off;
        off += (bytes + 255) & ~(size_t)255;
        return p;
    };
    __bf16* feat_bf = (__bf16*)carve((size_t)NN * WP * WP * CC * 2);  // padded, halo = 0
    __bf16* t_bf    = (__bf16*)carve((size_t)NN * HH * WW * CC * 2);
    __bf16* wp3     = (__bf16*)carve((size_t)9 * 8 * 16 * 512 * 2);
    __bf16* wph     = (__bf16*)carve((size_t)8 * 3 * 512 * 2);
    float* boxreg   = (float*)carve((size_t)NN * HWA * 4 * 4);
    float* obj      = (float*)carve((size_t)NN * HWA * 4);
    float* cand_s   = (float*)carve((size_t)NN * CAND_CAP * 4);
    int*   cand_i   = (int*)carve((size_t)NN * CAND_CAP * 4);
    int*   cand_cnt = (int*)carve((size_t)NN * 4);
    float* prop     = (float*)carve((size_t)NN * PRE_NMS * 4 * 4);
    float* pscore   = (float*)carve((size_t)NN * PRE_NMS * 4);
    int*   pvalid   = (int*)carve((size_t)NN * PRE_NMS * 4);

    float* out_props  = (float*)d_out;
    float* out_scrs   = out_props + NN * POST_NMS * 4;
    float* out_mask   = out_scrs + NN * POST_NMS;
    float* out_logits = out_mask + NN * POST_NMS;

    hipMemsetAsync(cand_cnt, 0, NN * sizeof(int), stream);

    const int padN = NN * WP * WP * CC;
    pack_feat_pad<<<(padN + 255) / 256, 256, 0, stream>>>(features, feat_bf, padN);
    pack_w3<<<(9 * 8 * 16 * 512 + 255) / 256, 256, 0, stream>>>(conv_w, wp3, 9 * 8 * 16 * 512);
    pack_whead<<<(8 * 3 * 512 + 255) / 256, 256, 0, stream>>>(cls_w, bbox_w, wph, 8 * 3 * 512);

    conv3x3_wmma<<<dim3(1024, 4), 256, 0, stream>>>(feat_bf, wp3, conv_b, t_bf);
    head_wmma<<<256, 256, 0, stream>>>(t_bf, wph, cls_b, bbox_b, out_logits, obj, boxreg);

    topk_select<<<NN, 256, 0, stream>>>(obj, cand_s, cand_i, cand_cnt);
    sort_cand<<<NN, 1024, 0, stream>>>(cand_s, cand_i, cand_cnt);
    decode_clip<<<(NN * PRE_NMS + 255) / 256, 256, 0, stream>>>(cand_s, cand_i, boxreg,
                                                                img_h, img_w, prop, pscore, pvalid);
    nms_kernel<<<NN, 256, 0, stream>>>(prop, pscore, pvalid, out_props, out_scrs, out_mask);
}
